// QRNNModel_84774064488822
// MI455X (gfx1250) — matrix operand order repaired and checked by reading, not compiled, not run on hardware
//
#include <hip/hip_runtime.h>

// QRNN (5 layers, S=1024 B=64 H=384) for MI455X / gfx1250.
// f16 WMMA GEMM with 4x4 register blocking (64x64 macro-tile per wave32,
// 1.0 loads per WMMA), operands L2-resident (50MB activations fit in the
// 192MB L2, so no LDS tiling), fused bias+activation epilogue writing f16
// gates, then a channel-parallel sequential scan (24576 independent
// recurrences, one lane each) for the fo_pool.

#define SEQ    1024
#define BATCH  64
#define DIN    384
#define HID    384
#define LAYERS 5
#define ROWS   (SEQ * BATCH)   // 65536
#define NCOL   (3 * HID)       // 1152

#define MT     4               // 16-row sub-tiles per wave
#define NT     4               // 16-col sub-tiles per wave
#define NT64   (NCOL / (16 * NT))   // 18 macro-tiles along N

typedef _Float16 half_t;
typedef _Float16 v16h __attribute__((ext_vector_type(16)));
typedef _Float16 v8h  __attribute__((ext_vector_type(8)));
typedef float    v8f  __attribute__((ext_vector_type(8)));

__device__ __forceinline__ float fast_sigmoid(float x) {
    float e = __expf(-x);                         // v_exp_f32 path
    return __builtin_amdgcn_rcpf(1.0f + e);       // v_rcp_f32
}
__device__ __forceinline__ float fast_tanh(float x) {
    // tanh(x) = 2/(1+exp(-2x)) - 1
    float e = __expf(-2.0f * x);
    return 2.0f * __builtin_amdgcn_rcpf(1.0f + e) - 1.0f;
}

// ---------------------------------------------------------------- converters
__global__ void cvt_x_kernel(const float* __restrict__ x,
                             half_t* __restrict__ h16, int n) {
    int i = blockIdx.x * blockDim.x + threadIdx.x;
    if (i < n) h16[i] = (half_t)x[i];
}

// W: (LAYERS, DIN, NCOL) row-major fp32  ->  Wt: (LAYERS, NCOL, DIN) f16
// (column-major per layer, so each WMMA B-lane reads two runs of 8 contiguous K halves)
__global__ void cvt_w_kernel(const float* __restrict__ W,
                             half_t* __restrict__ Wt, int n) {
    int i = blockIdx.x * blockDim.x + threadIdx.x;
    if (i >= n) return;
    int l = i / (NCOL * DIN);
    int r = i - l * (NCOL * DIN);
    int col = r / DIN;        // 0..1151
    int k   = r - col * DIN;  // 0..383
    Wt[i] = (half_t)W[(size_t)l * DIN * NCOL + (size_t)k * NCOL + col];
}

// ------------------------------------------------------------ GEMM + gates
// One wave32 per 64x64 output macro-tile of Y = h @ W + b, K = 384.
// 12 k-chunks x 16 WMMAs = 192 WMMAs per wave, 16 b128 loads per chunk.
// Epilogue applies tanh (Z block) or sigmoid (F/O blocks) and stores f16.
__global__ __launch_bounds__(256) void gemm_act_kernel(
    const half_t* __restrict__ h16,   // ROWS x DIN   (f16, row-major)
    const half_t* __restrict__ Wt,    // NCOL x DIN   (f16, W column-major)
    const float*  __restrict__ bias,  // NCOL
    half_t*       __restrict__ Yact)  // ROWS x NCOL  (f16 gates)
{
    const int lane   = threadIdx.x & 31;
    const int wave   = threadIdx.x >> 5;
    const int tile   = blockIdx.x * 8 + wave;      // 18432 macro-tiles total
    const int mt64   = tile / NT64;                // 0..1023
    const int nt64   = tile - mt64 * NT64;         // 0..17
    const int lane16 = lane & 15;
    const int laneHi = lane >> 4;

    // A-fragment (16x32 f16): lane<16 -> row m0+lane, K halves
    //   {k0..k0+7, k0+16..k0+23}; lane>=16 -> same rows, K shifted by +8.
    // Sub-tile sm adds a constant sm*16*DIN half offset (folds into imm).
    const half_t* Ap = h16 + (size_t)(mt64 * 64 + lane16) * DIN + laneHi * 8;
    // B-fragment (32x16 f16): lane<16 -> col n0+lane, K halves
    //   {k0..k0+7, k0+16..k0+23}; lane>=16 -> same cols, K shifted by +8.
    const half_t* Bp = Wt + (size_t)(nt64 * 64 + lane16) * DIN + laneHi * 8;

    v8f acc[MT][NT];
#pragma unroll
    for (int sm = 0; sm < MT; ++sm)
#pragma unroll
        for (int sn = 0; sn < NT; ++sn)
            acc[sm][sn] = (v8f){};

#pragma unroll
    for (int k0 = 0; k0 < DIN; k0 += 32) {
        union { v16h v; v8h h[2]; } a[MT], b[NT];
#pragma unroll
        for (int sm = 0; sm < MT; ++sm) {
            const half_t* p = Ap + (size_t)sm * 16 * DIN + k0;
            a[sm].h[0] = *(const v8h*)(p);
            a[sm].h[1] = *(const v8h*)(p + 16);
        }
#pragma unroll
        for (int sn = 0; sn < NT; ++sn) {
            const half_t* p = Bp + (size_t)sn * 16 * DIN + k0;
            b[sn].h[0] = *(const v8h*)(p);
            b[sn].h[1] = *(const v8h*)(p + 16);
        }
#pragma unroll
        for (int sm = 0; sm < MT; ++sm)
#pragma unroll
            for (int sn = 0; sn < NT; ++sn)
                acc[sm][sn] = __builtin_amdgcn_wmma_f32_16x16x32_f16(
                    /*neg_a=*/false, a[sm].v, /*neg_b=*/false, b[sn].v,
                    /*c_mod=*/(short)0, acc[sm][sn],
                    /*reuse_a=*/false, /*reuse_b=*/false);
    }

    // C/D layout: VGPR i -> row (laneHi?8:0)+i, col = lane16.
    // 64-col macro-tile lies entirely in one of Z/F/O (384 % 64 == 0).
    const int region = (nt64 * 64) / HID;          // 0=Z(tanh) 1=F 2=O (sigmoid)
#pragma unroll
    for (int sn = 0; sn < NT; ++sn) {
        const int   outCol = nt64 * 64 + sn * 16 + lane16;
        const float bia    = bias[outCol];
#pragma unroll
        for (int sm = 0; sm < MT; ++sm) {
            const int rowBase = mt64 * 64 + sm * 16 + laneHi * 8;
#pragma unroll
            for (int i = 0; i < 8; ++i) {
                float y = acc[sm][sn][i] + bia;
                float v = (region == 0) ? fast_tanh(y) : fast_sigmoid(y);
                Yact[(size_t)(rowBase + i) * NCOL + outCol] = (half_t)v;
            }
        }
    }
}

// --------------------------------------------------------------- fo_pool
// c_t = (1-F)c_{t-1} + F*Z = c + F*(Z-c);  out = sigmoid(O) * c.
// One lane per (batch, hidden) channel: 24576 lanes, 1024 serial steps each.
__global__ __launch_bounds__(256) void scan_kernel(
    const half_t* __restrict__ Yact,   // ROWS x NCOL gates
    half_t*       __restrict__ hNext,  // ROWS x HID  (next layer input, f16)
    float*        __restrict__ outF,   // final fp32 output (used when isLast)
    int isLast)
{
    const int ch = blockIdx.x * blockDim.x + threadIdx.x;  // 0..24575
    const int b  = ch / HID;
    const int j  = ch - b * HID;

    size_t yofs = (size_t)b * NCOL + j;
    size_t hofs = (size_t)b * HID + j;
    const size_t ystride = (size_t)BATCH * NCOL;
    const size_t hstride = (size_t)BATCH * HID;

    float c = 0.0f;
    for (int t = 0; t < SEQ; ++t) {
        // stream ahead: speculative prefetch (global_prefetch_b8), drops if OOB
        __builtin_prefetch(Yact + yofs + 8 * ystride, 0, 1);
        float Z = (float)Yact[yofs];
        float F = (float)Yact[yofs + HID];
        float O = (float)Yact[yofs + 2 * HID];
        c = fmaf(F, Z - c, c);
        float o = O * c;
        hNext[hofs] = (half_t)o;
        if (isLast) outF[hofs] = o;
        yofs += ystride;
        hofs += hstride;
    }
}

// ----------------------------------------------------------------- launcher
extern "C" void kernel_launch(void* const* d_in, const int* in_sizes, int n_in,
                              void* d_out, int out_size, void* d_ws, size_t ws_size,
                              hipStream_t stream) {
    (void)in_sizes; (void)n_in; (void)out_size; (void)ws_size;

    const float* x  = (const float*)d_in[0];   // (S,B,IN) fp32
    const float* Ws = (const float*)d_in[1];   // (L,IN,3H) fp32
    const float* bs = (const float*)d_in[2];   // (L,3H) fp32
    float* out = (float*)d_out;                // (S,B,H) fp32

    // Workspace layout (256B aligned):
    //   h16  : ROWS*DIN  f16  ( 50.3 MB)  - current layer input (in-place ping)
    //   Wt   : L*NCOL*DIN f16 (  4.4 MB)  - pre-transposed weights
    //   Yact : ROWS*NCOL f16  (151.0 MB)  - activated gates for current layer
    char* ws = (char*)d_ws;
    size_t off = 0;
    half_t* h16 = (half_t*)(ws + off);
    off += (size_t)ROWS * DIN * sizeof(half_t);  off = (off + 255) & ~(size_t)255;
    half_t* Wt = (half_t*)(ws + off);
    off += (size_t)LAYERS * NCOL * DIN * sizeof(half_t);  off = (off + 255) & ~(size_t)255;
    half_t* Yact = (half_t*)(ws + off);

    const int nX = ROWS * DIN;                 // 25,165,824
    const int nW = LAYERS * NCOL * DIN;        //  2,211,840
    cvt_x_kernel<<<(nX + 255) / 256, 256, 0, stream>>>(x, h16, nX);
    cvt_w_kernel<<<(nW + 255) / 256, 256, 0, stream>>>(Ws, Wt, nW);

    const int nMacro = (ROWS / 64) * NT64;     // 1024 * 18 = 18432, 8 waves/blk
    for (int l = 0; l < LAYERS; ++l) {
        gemm_act_kernel<<<nMacro / 8, 256, 0, stream>>>(
            h16, Wt + (size_t)l * NCOL * DIN, bs + (size_t)l * NCOL, Yact);
        scan_kernel<<<(BATCH * HID) / 256, 256, 0, stream>>>(
            Yact, h16, out, (l == LAYERS - 1) ? 1 : 0);
    }
}
